// LayerNormLSTM_33268816675345
// MI455X (gfx1250) — matrix operand order repaired and checked by reading, not compiled, not run on hardware
//
#include <hip/hip_runtime.h>
#include <hip/hip_bf16.h>

// ---------------------------------------------------------------------------
// LayerNorm-LSTM for MI455X (gfx1250, wave32, WMMA)
// B=64, S=512, I=512, H=1024, V=4096, G=4H=4096
// ---------------------------------------------------------------------------

typedef __attribute__((ext_vector_type(16))) __bf16 v16bf;
typedef __attribute__((ext_vector_type(8)))  float  v8f;

#define BB   64
#define SS   512
#define II   512
#define HH   1024
#define GG   4096   // 4*H
#define VV   4096
#define EPS  1e-5f

// ------------------------------ fp32 -> bf16 -------------------------------
__global__ void cvt_bf16_kernel(const float* __restrict__ in,
                                __hip_bfloat16* __restrict__ out, int n) {
    int i = (blockIdx.x * blockDim.x + threadIdx.x) * 4;
    if (i + 3 < n) {
        float4 f = *(const float4*)(in + i);
        out[i + 0] = __float2bfloat16(f.x);
        out[i + 1] = __float2bfloat16(f.y);
        out[i + 2] = __float2bfloat16(f.z);
        out[i + 3] = __float2bfloat16(f.w);
    } else {
        for (; i < n; ++i) out[i] = __float2bfloat16(in[i]);
    }
}

// ------------------- register-macro-tiled bf16 WMMA GEMM -------------------
// C[M,N] = A[M,K] * W[N,K]^T + bias[N]
// Each wave computes a (MT*16) x (NT*16) block of C. Fragment loads are
// software-pipelined x2 with *named* buffers (constant indexing only).
// Tile sizes kept small enough (MT*NT acc + 2*(MT+NT) frags) that everything
// stays in VGPRs -- no spills. K must be a multiple of 64.
// a_perm: logical row r (= s*64+b) reads A row (b*512 + s)    [x is (B,S,I)]
// c_perm: logical row r (= s*64+b) writes C[b*512*N + s*N + n] [y is (B,S,V)]
// c_bf16: store output as bf16 instead of fp32
template <int MT, int NT>
__global__ void gemm_wmma_bf16_kernel(const __hip_bfloat16* __restrict__ A,
                                      const __hip_bfloat16* __restrict__ W,
                                      const float* __restrict__ bias,
                                      void* __restrict__ Cout,
                                      int M, int N, int K,
                                      int a_perm, int c_perm, int c_bf16) {
    const int wave = (blockIdx.x * blockDim.x + threadIdx.x) >> 5;
    const int lane = threadIdx.x & 31;
    const int ntn  = N / (16 * NT);
    const int total = (M / (16 * MT)) * ntn;
    if (wave >= total) return;                 // wave-uniform, EXEC stays all-1s

    const int tm   = wave / ntn;               // macro-tile row index
    const int tn   = wave - tm * ntn;          // macro-tile col index
    const int half = lane >> 4;                // 0: lanes 0-15, 1: lanes 16-31
    const int l15  = lane & 15;
    const int akb  = half << 3;                // A frag K base: 0 or 8
    const int bkb  = half << 3;                // B frag K base: 0 or 8

    const __hip_bfloat16* Ap[MT];
    #pragma unroll
    for (int mt = 0; mt < MT; ++mt) {
        const int r_log = (tm * MT + mt) * 16 + l15;
        const int arow  = a_perm ? ((r_log & 63) * SS + (r_log >> 6)) : r_log;
        Ap[mt] = A + (size_t)arow * K;
    }
    const __hip_bfloat16* Wp[NT];
    #pragma unroll
    for (int nt = 0; nt < NT; ++nt)
        Wp[nt] = W + (size_t)((tn * NT + nt) * 16 + l15) * K;

    v8f acc[MT][NT];
    #pragma unroll
    for (int mt = 0; mt < MT; ++mt)
        #pragma unroll
        for (int nt = 0; nt < NT; ++nt) acc[mt][nt] = (v8f){};

    union Frag { v16bf v; uint4 q[2]; };
    Frag a0[MT], b0[NT], a1[MT], b1[NT];

    // A 16x32 bf16: lane holds K = akb..akb+7 then akb+16..akb+23
    // B 32x16 bf16: lane holds K = bkb..bkb+7 then bkb+16..bkb+23
    auto load_frags = [&](int kk, Frag* a, Frag* b) {
        #pragma unroll
        for (int mt = 0; mt < MT; ++mt) {
            a[mt].q[0] = *(const uint4*)(Ap[mt] + kk + akb);
            a[mt].q[1] = *(const uint4*)(Ap[mt] + kk + akb + 16);
            // keep the streaming A operand ahead in GL2
            __builtin_prefetch((const void*)(Ap[mt] + kk + 192), 0, 3);
        }
        #pragma unroll
        for (int nt = 0; nt < NT; ++nt) {
            b[nt].q[0] = *(const uint4*)(Wp[nt] + kk + bkb);
            b[nt].q[1] = *(const uint4*)(Wp[nt] + kk + bkb + 16);
        }
    };
    auto do_wmma = [&](Frag* a, Frag* b) {
        #pragma unroll
        for (int mt = 0; mt < MT; ++mt)
            #pragma unroll
            for (int nt = 0; nt < NT; ++nt)
                acc[mt][nt] = __builtin_amdgcn_wmma_f32_16x16x32_bf16(
                    false, a[mt].v, false, b[nt].v, (short)0, acc[mt][nt],
                    false, false);
    };

    load_frags(0, a0, b0);
    for (int kk = 0; kk < K; kk += 64) {
        load_frags(kk + 32, a1, b1);           // in flight over these WMMAs
        do_wmma(a0, b0);
        if (kk + 64 < K) load_frags(kk + 64, a0, b0);
        do_wmma(a1, b1);
    }

    #pragma unroll
    for (int nt = 0; nt < NT; ++nt) {
        const int col = (tn * NT + nt) * 16 + l15;
        const float bv = bias ? bias[col] : 0.0f;
        #pragma unroll
        for (int mt = 0; mt < MT; ++mt) {
            #pragma unroll
            for (int r = 0; r < 8; ++r) {
                const int row = (tm * MT + mt) * 16 + r + (half ? 8 : 0);
                const float val = acc[mt][nt][r] + bv;
                size_t idx;
                if (c_perm) {
                    const int b = row & 63, s = row >> 6;
                    idx = (size_t)b * ((size_t)SS * N) + (size_t)s * N + col;
                } else {
                    idx = (size_t)row * N + col;
                }
                if (c_bf16) ((__hip_bfloat16*)Cout)[idx] = __float2bfloat16(val);
                else        ((float*)Cout)[idx] = val;
            }
        }
    }
}

// --------------------------- block reductions ------------------------------
// 4 simultaneous block sums with a single barrier sequence (wave32 shuffles
// + one LDS round-trip). blockDim.x must be a multiple of 32, <= 1024.
__device__ __forceinline__ float4 block_reduce_sum4(float v0, float v1,
                                                    float v2, float v3,
                                                    float4* smem) {
    #pragma unroll
    for (int off = 16; off > 0; off >>= 1) {
        v0 += __shfl_down(v0, off, 32);
        v1 += __shfl_down(v1, off, 32);
        v2 += __shfl_down(v2, off, 32);
        v3 += __shfl_down(v3, off, 32);
    }
    const int lane = threadIdx.x & 31;
    const int wid  = threadIdx.x >> 5;
    if (lane == 0) smem[wid] = make_float4(v0, v1, v2, v3);
    __syncthreads();
    if (wid == 0) {
        const int nw = blockDim.x >> 5;
        float4 t = (lane < nw) ? smem[lane] : make_float4(0.f, 0.f, 0.f, 0.f);
        #pragma unroll
        for (int off = 16; off > 0; off >>= 1) {
            t.x += __shfl_down(t.x, off, 32);
            t.y += __shfl_down(t.y, off, 32);
            t.z += __shfl_down(t.z, off, 32);
            t.w += __shfl_down(t.w, off, 32);
        }
        if (lane == 0) smem[0] = t;
    }
    __syncthreads();
    const float4 r = smem[0];
    __syncthreads();
    return r;
}

__device__ __forceinline__ float sigmoidf(float x) {
    return 1.0f / (1.0f + __expf(-x));
}

// ---------------------- fused LN + gates + cell/hidden ---------------------
// grid = B blocks, block = H threads. Thread k handles gate cols k + j*H.
__global__ void lstm_step_kernel(const __hip_bfloat16* __restrict__ xs_t, // [B,4H]
                                 const float* __restrict__ hh,            // [B,4H]
                                 const float* __restrict__ g_ih,
                                 const float* __restrict__ be_ih,
                                 const float* __restrict__ g_hh,
                                 const float* __restrict__ be_hh,
                                 const float* __restrict__ g_ho,
                                 const float* __restrict__ be_ho,
                                 float* __restrict__ c,                   // [B,H]
                                 __hip_bfloat16* __restrict__ h,          // [B,H]
                                 __hip_bfloat16* __restrict__ hs_t) {     // [B,H]
    __shared__ float4 smem[32];
    const int b = blockIdx.x;
    const int k = threadIdx.x;

    const __hip_bfloat16* xr = xs_t + (size_t)b * GG;
    const float*          hr = hh   + (size_t)b * GG;

    float xv[4], hv[4];
    float sx = 0.f, sx2 = 0.f, sh = 0.f, sh2 = 0.f;
    #pragma unroll
    for (int j = 0; j < 4; ++j) {
        const int col = k + j * HH;
        xv[j] = __bfloat162float(xr[col]);
        hv[j] = hr[col];
        sx += xv[j]; sx2 += xv[j] * xv[j];
        sh += hv[j]; sh2 += hv[j] * hv[j];
    }
    const float4 s4 = block_reduce_sum4(sx, sx2, sh, sh2, smem);
    const float mx = s4.x * (1.0f / GG);
    const float mh = s4.z * (1.0f / GG);
    const float rx = rsqrtf(s4.y * (1.0f / GG) - mx * mx + EPS);
    const float rh = rsqrtf(s4.w * (1.0f / GG) - mh * mh + EPS);

    float gate[4];
    #pragma unroll
    for (int j = 0; j < 4; ++j) {
        const int col = k + j * HH;
        const float gx = (xv[j] - mx) * rx * g_ih[col] + be_ih[col];
        const float gh = (hv[j] - mh) * rh * g_hh[col] + be_hh[col];
        gate[j] = gx + gh;
    }
    const float ig = sigmoidf(gate[0]);
    const float fg = sigmoidf(gate[1]);
    const float og = sigmoidf(gate[2]);
    const float gg = tanhf(gate[3]);

    const size_t ci = (size_t)b * HH + k;
    const float cn = fg * c[ci] + ig * gg;
    c[ci] = cn;

    const float4 c4 = block_reduce_sum4(cn, cn * cn, 0.f, 0.f, smem);
    const float mc = c4.x * (1.0f / HH);
    const float rc = rsqrtf(c4.y * (1.0f / HH) - mc * mc + EPS);

    const float hn = og * tanhf((cn - mc) * rc * g_ho[k] + be_ho[k]);
    const __hip_bfloat16 hb = __float2bfloat16(hn);
    h[ci]    = hb;
    hs_t[ci] = hb;
}

// ------------------------------ host driver --------------------------------
extern "C" void kernel_launch(void* const* d_in, const int* in_sizes, int n_in,
                              void* d_out, int out_size, void* d_ws, size_t ws_size,
                              hipStream_t stream) {
    const float* x     = (const float*)d_in[0];
    const float* W_ih  = (const float*)d_in[1];
    const float* b_ih  = (const float*)d_in[2];
    const float* W_hh  = (const float*)d_in[3];
    const float* b_hh  = (const float*)d_in[4];
    const float* g_ih  = (const float*)d_in[5];
    const float* be_ih = (const float*)d_in[6];
    const float* g_hh  = (const float*)d_in[7];
    const float* be_hh = (const float*)d_in[8];
    const float* g_ho  = (const float*)d_in[9];
    const float* be_ho = (const float*)d_in[10];
    const float* W_out = (const float*)d_in[11];
    const float* b_out = (const float*)d_in[12];
    float* y = (float*)d_out;

    char* ws = (char*)d_ws;
    size_t off = 0;
    auto alloc = [&](size_t bytes) -> char* {
        char* p = ws + off;
        off += (bytes + 255) & ~(size_t)255;
        return p;
    };
    __hip_bfloat16* x_bf    = (__hip_bfloat16*)alloc((size_t)BB * SS * II * 2);
    __hip_bfloat16* Wih_bf  = (__hip_bfloat16*)alloc((size_t)GG * II * 2);
    __hip_bfloat16* Whh_bf  = (__hip_bfloat16*)alloc((size_t)GG * HH * 2);
    __hip_bfloat16* Wout_bf = (__hip_bfloat16*)alloc((size_t)VV * HH * 2);
    __hip_bfloat16* xs_bf   = (__hip_bfloat16*)alloc((size_t)SS * BB * GG * 2);
    __hip_bfloat16* hs_bf   = (__hip_bfloat16*)alloc((size_t)SS * BB * HH * 2);
    __hip_bfloat16* h_bf    = (__hip_bfloat16*)alloc((size_t)BB * HH * 2);
    float*          hh_f    = (float*)alloc((size_t)BB * GG * 4);
    float*          c_f     = (float*)alloc((size_t)BB * HH * 4);

    auto cvt = [&](const float* in, __hip_bfloat16* out, size_t n) {
        int grid = (int)((n / 4 + 255) / 256);
        cvt_bf16_kernel<<<grid, 256, 0, stream>>>(in, out, (int)n);
    };
    cvt(x,     x_bf,    (size_t)BB * SS * II);
    cvt(W_ih,  Wih_bf,  (size_t)GG * II);
    cvt(W_hh,  Whh_bf,  (size_t)GG * HH);
    cvt(W_out, Wout_bf, (size_t)VV * HH);

    // Big GEMMs: 32x32 macro-tile per wave, x2-pipelined, spill-free
    // (4 WMMA per 4 in-flight fragments; ~120 VGPRs total).
    auto gemm_big = [&](const __hip_bfloat16* A, const __hip_bfloat16* W,
                        const float* bias, void* C, int M, int N, int K,
                        int ap, int cp, int cb) {
        long long waves = (long long)(M / 32) * (N / 32);
        int grid = (int)((waves * 32 + 255) / 256);
        gemm_wmma_bf16_kernel<2, 2><<<grid, 256, 0, stream>>>(A, W, bias, C,
                                                              M, N, K, ap, cp, cb);
    };
    // Recurrent GEMM (M=64): 16x64 macro-tile keeps 256 waves in flight.
    auto gemm_step = [&](const __hip_bfloat16* A, const __hip_bfloat16* W,
                         const float* bias, void* C, int M, int N, int K) {
        long long waves = (long long)(M / 16) * (N / 64);
        int grid = (int)((waves * 32 + 255) / 256);
        gemm_wmma_bf16_kernel<1, 4><<<grid, 256, 0, stream>>>(A, W, bias, C,
                                                              M, N, K, 0, 0, 0);
    };

    // xs[s,b,:] = x[b,s,:] @ W_ih^T + b_ih    (stored bf16, 256 MB)
    gemm_big(x_bf, Wih_bf, b_ih, xs_bf, BB * SS, GG, II, /*ap=*/1, /*cp=*/0, /*cb=*/1);

    // h0 = 0, c0 = 0
    hipMemsetAsync(h_bf, 0, (size_t)BB * HH * 2, stream);
    hipMemsetAsync(c_f,  0, (size_t)BB * HH * 4, stream);

    // recurrent scan: 512 sequential (GEMM -> fused LN/gates) pairs
    for (int t = 0; t < SS; ++t) {
        gemm_step(h_bf, Whh_bf, b_hh, hh_f, BB, GG, HH);
        lstm_step_kernel<<<BB, HH, 0, stream>>>(
            xs_bf + (size_t)t * BB * GG, hh_f,
            g_ih, be_ih, g_hh, be_hh, g_ho, be_ho,
            c_f, h_bf, hs_bf + (size_t)t * BB * HH);
    }

    // y[b,s,:] = hs[s,b,:] @ W_out^T + b_out  (fp32 out, permuted store)
    gemm_big(hs_bf, Wout_bf, b_out, y, BB * SS, VV, HH, /*ap=*/0, /*cp=*/1, /*cb=*/0);
}
